// LSTM_74543452389916
// MI455X (gfx1250) — matrix operand order repaired and checked by reading, not compiled, not run on hardware
//
#include <hip/hip_runtime.h>
#include <hip/hip_bf16.h>
#include <stdint.h>

typedef _Float16 v16h __attribute__((ext_vector_type(16)));
typedef _Float16 v8h  __attribute__((ext_vector_type(8)));
typedef float    v8f  __attribute__((ext_vector_type(8)));

#define SEQ   256
#define HID   512
#define OUTN  128
#define BATCH 512

// LDS row stride in halfs: 88 halfs = 176B (16B aligned rows, stride/4=44 word
// stride -> gcd(44,64)=4 so 16 distinct bank groups across a 16-lane ds_load_b128).
#define LDSK  88

// ---- workspace layout (bytes) ----
static const size_t OFF_WH   = 0;                              // f16 [4][HID][HID]
static const size_t SZ_WH    = (size_t)4 * HID * HID * 2;
static const size_t OFF_WP   = OFF_WH + SZ_WH;                 // f16 [OUTN][HID]
static const size_t SZ_WP    = (size_t)OUTN * HID * 2;
static const size_t OFF_WX   = OFF_WP + SZ_WP;                 // f32 [4][HID]
static const size_t SZ_WX    = (size_t)4 * HID * 4;
static const size_t OFF_BIAS = OFF_WX + SZ_WX;                 // f32 [4][HID] (bx+bh)
static const size_t SZ_BIAS  = (size_t)4 * HID * 4;
static const size_t OFF_H0   = OFF_BIAS + SZ_BIAS;             // f16 [BATCH][HID]
static const size_t SZ_H     = (size_t)BATCH * HID * 2;
static const size_t OFF_H1   = OFF_H0 + SZ_H;                  // f16 [BATCH][HID]
static const size_t OFF_C    = OFF_H1 + SZ_H;                  // f32 [BATCH][HID]

__device__ __forceinline__ float sigmoidf_(float z) {
    return 1.0f / (1.0f + __expf(-z));
}

// async DMA of one 16-byte chunk: global -> LDS (ASYNCcnt-tracked, no VGPR data)
__device__ __forceinline__ void async_copy16(const _Float16* gsrc, _Float16* lds_dst) {
    uint32_t loff = (uint32_t)(uintptr_t)lds_dst;   // generic LDS ptr: addr[31:0] = LDS byte offset
    asm volatile("global_load_async_to_lds_b128 %0, %1, off"
                 :: "v"(loff), "v"(gsrc) : "memory");
}

__device__ __forceinline__ void wait_async0() {
    asm volatile("s_wait_asynccnt 0x0" ::: "memory");
}

// ---------------- weight packing ----------------
__global__ __launch_bounds__(256) void lstm_pack_kernel(
    const float* __restrict__ Wgh, const float* __restrict__ Wih,
    const float* __restrict__ Wfh, const float* __restrict__ Woh,
    const float* __restrict__ Wgx, const float* __restrict__ Wix,
    const float* __restrict__ Wfx, const float* __restrict__ Wox,
    const float* __restrict__ bgx, const float* __restrict__ bix,
    const float* __restrict__ bfx, const float* __restrict__ box,
    const float* __restrict__ bgh, const float* __restrict__ bih,
    const float* __restrict__ bfh, const float* __restrict__ boh,
    const float* __restrict__ Wp,
    _Float16* __restrict__ WhP, _Float16* __restrict__ WpP,
    float* __restrict__ wx, float* __restrict__ bias)
{
    const size_t NWh = (size_t)4 * HID * HID;
    const size_t NWp = (size_t)OUTN * HID;
    size_t tid = (size_t)blockIdx.x * blockDim.x + threadIdx.x;
    if (tid < NWh) {
        int gate = (int)(tid / ((size_t)HID * HID));
        size_t idx = tid % ((size_t)HID * HID);
        const float* W = (gate == 0) ? Wgh : (gate == 1) ? Wih : (gate == 2) ? Wfh : Woh;
        WhP[tid] = (_Float16)W[idx];
    } else if (tid < NWh + NWp) {
        size_t i = tid - NWh;
        WpP[i] = (_Float16)Wp[i];
    } else if (tid < NWh + NWp + (size_t)4 * HID) {
        size_t i = tid - NWh - NWp;
        int gate = (int)(i / HID);
        int j = (int)(i % HID);
        const float* Wxp = (gate == 0) ? Wgx : (gate == 1) ? Wix : (gate == 2) ? Wfx : Wox;
        const float* bxp = (gate == 0) ? bgx : (gate == 1) ? bix : (gate == 2) ? bfx : box;
        const float* bhp = (gate == 0) ? bgh : (gate == 1) ? bih : (gate == 2) ? bfh : boh;
        wx[i]   = Wxp[j];
        bias[i] = bxp[j] + bhp[j];
    }
}

__global__ __launch_bounds__(256) void lstm_zero_kernel(uint32_t* __restrict__ p, size_t n) {
    size_t i = (size_t)blockIdx.x * blockDim.x + threadIdx.x;
    if (i < n) p[i] = 0u;
}

// ---------------- one LSTM timestep ----------------
// grid (BATCH/64, HID/64), 256 threads = 8 waves (wave32).
// K processed in chunks of 64: async-DMA the h-tile chunk (64x64) and the
// weight chunk (4 gates x 64 j x 64 k) into LDS, s_wait_asynccnt, barrier,
// then two 16x16x32 WMMA k-steps per wave fed from ds_load_b128.
// Each wave computes all 4 gates for its 16x32 patch -> wave-local c/h update.
__global__ __launch_bounds__(256) void lstm_step_kernel(
    const float*    __restrict__ x,     // [BATCH][SEQ]
    const _Float16* __restrict__ WhP,   // [4][HID][HID]
    const float*    __restrict__ wx,    // [4][HID]
    const float*    __restrict__ bias,  // [4][HID]
    const _Float16* __restrict__ hin,   // [BATCH][HID]
    _Float16*       __restrict__ hout,  // [BATCH][HID]
    float*          __restrict__ cbuf,  // [BATCH][HID]
    int t)
{
    __shared__ __align__(16) _Float16 As[64 * LDSK];        // 11,264 B
    __shared__ __align__(16) _Float16 Bs[4 * 64 * LDSK];    // 45,056 B

    const int tid  = threadIdx.x;
    const int wave = tid >> 5;
    const int lane = tid & 31;
    const int mw = wave & 3;
    const int nw = wave >> 2;
    const int mB = blockIdx.x * 64;              // block batch-row base
    const int jB = blockIdx.y * 64;              // block hidden-col base
    const int m0 = mB + mw * 16;                 // wave's 16 batch rows
    const int j0 = jB + nw * 32;                 // wave's 32 hidden cols
    const int lrow = lane & 15;
    const int lsel = lane >> 4;

    v8f acc[4][2] = {};

    for (int kc = 0; kc < HID; kc += 64) {
        __syncthreads();   // previous chunk's LDS reads finished

        // stage A chunk: 64 rows x 64 halfs = 512 x 16B; 2 chunks per thread
#pragma unroll
        for (int q = 0; q < 2; ++q) {
            int c = tid + q * 256;
            int row = c >> 3, part = c & 7;
            async_copy16(hin + (size_t)(mB + row) * HID + kc + part * 8,
                         &As[row * LDSK + part * 8]);
        }
        // stage B chunk: 4*64 rows x 64 halfs = 2048 x 16B; 8 chunks per thread
#pragma unroll
        for (int q = 0; q < 8; ++q) {
            int c = tid + q * 256;
            int rj = c >> 3, part = c & 7;
            int gate = rj >> 6, jl = rj & 63;
            async_copy16(WhP + ((size_t)gate * HID + jB + jl) * HID + kc + part * 8,
                         &Bs[(gate * 64 + jl) * LDSK + part * 8]);
        }
        wait_async0();
        __syncthreads();   // staged data visible to all waves

#pragma unroll
        for (int kk = 0; kk < 64; kk += 32) {
            v16h a;
            const _Float16* arow = &As[(mw * 16 + lrow) * LDSK + kk + lsel * 16];
            *reinterpret_cast<v8h*>(&a)       = *reinterpret_cast<const v8h*>(arow);
            *(reinterpret_cast<v8h*>(&a) + 1) = *reinterpret_cast<const v8h*>(arow + 8);
#pragma unroll
            for (int g = 0; g < 4; ++g) {
#pragma unroll
                for (int ni = 0; ni < 2; ++ni) {
                    const _Float16* brow =
                        &Bs[(g * 64 + nw * 32 + ni * 16 + lrow) * LDSK + kk + lsel * 16];
                    v16h b = *reinterpret_cast<const v16h*>(brow);
                    acc[g][ni] = __builtin_amdgcn_wmma_f32_16x16x32_f16(
                        false, a, false, b, (short)0, acc[g][ni], false, false);
                }
            }
        }
    }

    // per-lane (m,j): C/D layout -> VGPR r, lanes0-15: M=r, lanes16-31: M=r+8
    float wxv[2][4], bv[2][4];
#pragma unroll
    for (int ni = 0; ni < 2; ++ni) {
        int j = j0 + ni * 16 + lrow;
#pragma unroll
        for (int g = 0; g < 4; ++g) {
            wxv[ni][g] = wx[g * HID + j];
            bv[ni][g]  = bias[g * HID + j];
        }
    }

#pragma unroll
    for (int r = 0; r < 8; ++r) {
        int m = m0 + lsel * 8 + r;
        float xv = x[(size_t)m * SEQ + t];
#pragma unroll
        for (int ni = 0; ni < 2; ++ni) {
            int j = j0 + ni * 16 + lrow;
            float zg = acc[0][ni][r] + xv * wxv[ni][0] + bv[ni][0];
            float zi = acc[1][ni][r] + xv * wxv[ni][1] + bv[ni][1];
            float zf = acc[2][ni][r] + xv * wxv[ni][2] + bv[ni][2];
            float zo = acc[3][ni][r] + xv * wxv[ni][3] + bv[ni][3];
            float gg = tanhf(zg);
            float ii = sigmoidf_(zi);
            float ff = sigmoidf_(zf);
            float oo = sigmoidf_(zo);
            size_t cidx = (size_t)m * HID + j;
            float c = cbuf[cidx];
            c = c * ff + gg * ii;
            cbuf[cidx] = c;
            hout[cidx] = (_Float16)(tanhf(c) * oo);
        }
    }
}

// ---------------- final projection: logits = h @ Wp^T + bp ----------------
__global__ __launch_bounds__(256) void lstm_proj_kernel(
    const _Float16* __restrict__ hin,   // [BATCH][HID]
    const _Float16* __restrict__ WpP,   // [OUTN][HID]
    const float*    __restrict__ bp,    // [OUTN]
    float*          __restrict__ out)   // [BATCH][OUTN]
{
    const int wave = threadIdx.x >> 5;
    const int lane = threadIdx.x & 31;
    const int mw = wave & 3;
    const int nw = wave >> 2;
    const int m0 = blockIdx.x * 64 + mw * 16;
    const int j0 = blockIdx.y * 64 + nw * 32;
    const int lrow = lane & 15;
    const int lsel = lane >> 4;

    const _Float16* aptr = hin + (size_t)(m0 + lrow) * HID + lsel * 16;
    const _Float16* bptr[2];
#pragma unroll
    for (int ni = 0; ni < 2; ++ni)
        bptr[ni] = WpP + (size_t)(j0 + ni * 16 + lrow) * HID + lsel * 16;

    v8f acc[2] = {};
#pragma unroll 4
    for (int k0 = 0; k0 < HID; k0 += 32) {
        v16h a;
        *reinterpret_cast<v8h*>(&a)       = *reinterpret_cast<const v8h*>(aptr + k0);
        *(reinterpret_cast<v8h*>(&a) + 1) = *reinterpret_cast<const v8h*>(aptr + k0 + 8);
#pragma unroll
        for (int ni = 0; ni < 2; ++ni) {
            v16h b = *reinterpret_cast<const v16h*>(bptr[ni] + k0);
            acc[ni] = __builtin_amdgcn_wmma_f32_16x16x32_f16(
                false, a, false, b, (short)0, acc[ni], false, false);
        }
    }

#pragma unroll
    for (int ni = 0; ni < 2; ++ni) {
        int j = j0 + ni * 16 + lrow;
        float bb = bp[j];
#pragma unroll
        for (int r = 0; r < 8; ++r) {
            int m = m0 + lsel * 8 + r;
            out[(size_t)m * OUTN + j] = acc[ni][r] + bb;
        }
    }
}

// ---------------- softmax over rows of [BATCH][OUTN], one wave32 per row ----
__global__ __launch_bounds__(32) void lstm_softmax_kernel(float* __restrict__ out) {
    int row = blockIdx.x;
    int lane = threadIdx.x;
    float v[4];
    float mx = -3.0e38f;
#pragma unroll
    for (int q = 0; q < 4; ++q) {
        v[q] = out[(size_t)row * OUTN + lane + q * 32];
        mx = fmaxf(mx, v[q]);
    }
#pragma unroll
    for (int off = 16; off > 0; off >>= 1)
        mx = fmaxf(mx, __shfl_xor(mx, off, 32));
    float s = 0.0f;
#pragma unroll
    for (int q = 0; q < 4; ++q) {
        v[q] = __expf(v[q] - mx);
        s += v[q];
    }
#pragma unroll
    for (int off = 16; off > 0; off >>= 1)
        s += __shfl_xor(s, off, 32);
    float inv = 1.0f / s;
#pragma unroll
    for (int q = 0; q < 4; ++q)
        out[(size_t)row * OUTN + lane + q * 32] = v[q] * inv;
}

extern "C" void kernel_launch(void* const* d_in, const int* in_sizes, int n_in,
                              void* d_out, int out_size, void* d_ws, size_t ws_size,
                              hipStream_t stream) {
    (void)in_sizes; (void)n_in; (void)out_size; (void)ws_size;

    const float* x   = (const float*)d_in[0];
    const float* Wgx = (const float*)d_in[1];
    const float* bgx = (const float*)d_in[2];
    const float* Wgh = (const float*)d_in[3];
    const float* bgh = (const float*)d_in[4];
    const float* Wix = (const float*)d_in[5];
    const float* bix = (const float*)d_in[6];
    const float* Wih = (const float*)d_in[7];
    const float* bih = (const float*)d_in[8];
    const float* Wfx = (const float*)d_in[9];
    const float* bfx = (const float*)d_in[10];
    const float* Wfh = (const float*)d_in[11];
    const float* bfh = (const float*)d_in[12];
    const float* Wox = (const float*)d_in[13];
    const float* box_ = (const float*)d_in[14];
    const float* Woh = (const float*)d_in[15];
    const float* boh = (const float*)d_in[16];
    const float* Wp  = (const float*)d_in[17];
    const float* bp  = (const float*)d_in[18];

    char* ws = (char*)d_ws;
    _Float16* WhP  = (_Float16*)(ws + OFF_WH);
    _Float16* WpP  = (_Float16*)(ws + OFF_WP);
    float*    wx   = (float*)(ws + OFF_WX);
    float*    bias = (float*)(ws + OFF_BIAS);
    _Float16* h0   = (_Float16*)(ws + OFF_H0);
    _Float16* h1   = (_Float16*)(ws + OFF_H1);
    float*    cbuf = (float*)(ws + OFF_C);

    // pack weights (f32 -> f16) + fuse biases
    {
        size_t total = (size_t)4 * HID * HID + (size_t)OUTN * HID + (size_t)4 * HID;
        int blocks = (int)((total + 255) / 256);
        lstm_pack_kernel<<<blocks, 256, 0, stream>>>(
            Wgh, Wih, Wfh, Woh, Wgx, Wix, Wfx, Wox,
            bgx, bix, bfx, box_, bgh, bih, bfh, boh,
            Wp, WhP, WpP, wx, bias);
    }

    // zero h0, h1, c (contiguous region)
    {
        size_t words = ((size_t)2 * SZ_H + (size_t)BATCH * HID * 4) / 4;
        int blocks = (int)((words + 255) / 256);
        lstm_zero_kernel<<<blocks, 256, 0, stream>>>((uint32_t*)(ws + OFF_H0), words);
    }

    // 256 recurrent steps; separate launches give grid-wide ordering.
    // step t reads h[t%2], writes h[(t+1)%2]; final h lands in h0.
    dim3 sgrid(BATCH / 64, HID / 64);
    for (int t = 0; t < SEQ; ++t) {
        _Float16* hin  = (t & 1) ? h1 : h0;
        _Float16* hout = (t & 1) ? h0 : h1;
        lstm_step_kernel<<<sgrid, 256, 0, stream>>>(x, WhP, wx, bias, hin, hout, cbuf, t);
    }

    // projection + softmax
    dim3 pgrid(BATCH / 64, OUTN / 64);
    lstm_proj_kernel<<<pgrid, 256, 0, stream>>>(h0, WpP, bp, (float*)d_out);
    lstm_softmax_kernel<<<BATCH, 32, 0, stream>>>((float*)d_out);
}
